// GCNJK_62689342652840
// MI455X (gfx1250) — matrix operand order repaired and checked by reading, not compiled, not run on hardware
//
#include <hip/hip_runtime.h>
#include <hip/hip_bf16.h>
#include <cstdint>
#include <cstddef>

// ---------------------------------------------------------------------------
// GCN (3-layer) + BN + JK-max + linear head + log_softmax for MI455X (gfx1250)
// GEMMs: v_wmma_f32_16x16x32_bf16 (wave32 WMMA), A-tile staged with
// global_load_async_to_lds_b128 (ASYNCcnt). Aggregation uses f32 atomics that
// stay resident in the 192MB L2 (out buffer = 102MB).
// ---------------------------------------------------------------------------

typedef __bf16 bf16_t;
typedef __attribute__((ext_vector_type(16))) __bf16 v16bf;
typedef __attribute__((ext_vector_type(8)))  float  v8f;

#define HIDC 512
#define OUTC 40

__device__ __forceinline__ bf16_t f2bf(float f) {
    union { float f; unsigned u; } x; x.f = f;
    unsigned r = x.u + 0x7fffu + ((x.u >> 16) & 1u);     // round-to-nearest-even
    unsigned short h = (unsigned short)(r >> 16);
    union { unsigned short s; bf16_t b; } y; y.s = h;
    return y.b;
}

// CDNA5 async global->LDS copy (16B per lane), tracked by ASYNCcnt.
__device__ __forceinline__ void async_load_b128(unsigned lds_off, const void* gaddr) {
    asm volatile("global_load_async_to_lds_b128 %0, %1, off"
                 :: "v"(lds_off), "v"(gaddr)
                 : "memory");
}
__device__ __forceinline__ void wait_asynccnt0() {
    asm volatile("s_wait_asynccnt 0x0" ::: "memory");
}

// ----------------------------- small utility kernels -----------------------

__global__ void fill_f32_kernel(float* __restrict__ p, float v, size_t n) {
    size_t i = (size_t)blockIdx.x * blockDim.x + threadIdx.x;
    if (i < n) p[i] = v;
}

__global__ void cvt_bf16_kernel(const float* __restrict__ in, bf16_t* __restrict__ out, size_t n) {
    size_t i = (size_t)blockIdx.x * blockDim.x + threadIdx.x;
    if (i < n) out[i] = f2bf(in[i]);
}

__global__ void deg_kernel(const int* __restrict__ dst, float* __restrict__ deg, int E) {
    int i = blockIdx.x * blockDim.x + threadIdx.x;
    if (i < E) atomicAdd(&deg[dst[i]], 1.0f);
}

__global__ void rsqrt_kernel(float* __restrict__ p, int n) {
    int i = blockIdx.x * blockDim.x + threadIdx.x;
    if (i < n) p[i] = rsqrtf(fmaxf(p[i], 1e-12f));
}

// out[i] = bias[i % C]  (so segment_sum lands on top of the bias)
__global__ void bias_init_kernel(float* __restrict__ out, const float* __restrict__ b, size_t n) {
    size_t i = (size_t)blockIdx.x * blockDim.x + threadIdx.x;
    if (i < n) out[i] = b[i & (HIDC - 1)];
}

// ----------------------------- WMMA GEMM ------------------------------------
// C[M x 512] = A[M x 512](bf16) * B[512 x 512](bf16), fp32 accum.
// Block = 256 threads (8 waves), tile 128x128, wave tile 32x64 (2x4 WMMA accums).
// K and N strides are compile-time (512) so epilogue stores use immediate offsets.

#define TM 128
#define TN 128
#define TK 32
#define SAP 40   // padded LDS row stride (halves) for A
#define SBP 40   // padded LDS row stride (halves) for Bt

__global__ __launch_bounds__(256) void gemm_bf16_wmma_kernel(
    const bf16_t* __restrict__ A, const bf16_t* __restrict__ B,
    float* __restrict__ C, int M)
{
    __shared__ bf16_t Alds[TM * SAP];       // [row][k]            10.0 KB
    __shared__ bf16_t Blds[TN * SBP];       // [n][k] (transposed) 10.0 KB

    const int tid   = threadIdx.x;
    const int wave  = tid >> 5;
    const int lane  = tid & 31;
    const int lhalf = (lane >> 4) & 1;       // 0: lanes 0-15, 1: lanes 16-31
    const int l16   = lane & 15;

    const int m0 = blockIdx.y * TM;
    const int n0 = blockIdx.x * TN;
    const int wr = (wave & 3) * 32;          // wave row offset within tile
    const int wc = (wave >> 2) * 64;         // wave col offset within tile

    v8f acc[2][4];
    #pragma unroll
    for (int i = 0; i < 2; ++i)
        #pragma unroll
        for (int j = 0; j < 4; ++j) acc[i][j] = {};

    for (int k0 = 0; k0 < HIDC; k0 += TK) {
        __syncthreads();                     // previous compute done before overwrite
        // ---- stage A tile 128x32 via async global->LDS b128 (2 per thread) ----
        #pragma unroll
        for (int it = 0; it < 2; ++it) {
            int idx  = tid + it * 256;
            int row  = idx >> 2;             // 0..127
            int kc   = idx & 3;              // 4 x 8-half chunks per row
            int grow = m0 + row; if (grow >= M) grow = M - 1;   // clamp (stores guarded)
            unsigned lds_off = (unsigned)(uintptr_t)(&Alds[row * SAP + kc * 8]);
            async_load_b128(lds_off, A + (size_t)grow * HIDC + k0 + kc * 8);
        }
        // ---- stage B tile 32x128 transposed into Blds[n][k] ----
        #pragma unroll
        for (int it = 0; it < 2; ++it) {
            int idx = tid + it * 256;
            int kr  = idx >> 4;              // 0..31
            int nc  = idx & 15;              // chunk of 8 columns
            uint4 v = *(const uint4*)(B + (size_t)(k0 + kr) * HIDC + n0 + nc * 8);
            const bf16_t* pv = (const bf16_t*)&v;
            #pragma unroll
            for (int j = 0; j < 8; ++j)
                Blds[(size_t)(nc * 8 + j) * SBP + kr] = pv[j];
        }
        wait_asynccnt0();                    // my async A copies landed in LDS
        __syncthreads();                     // (compiler also waits DScnt for B)

        // A fragment (16-bit 16x32 layout): lanes 0-15 -> K[0..7],[16..23];
        //                                   lanes 16-31 -> K[8..15],[24..31]
        v16bf afrag[2];
        #pragma unroll
        for (int i = 0; i < 2; ++i) {
            int row = wr + i * 16 + l16;
            union { v16bf v; uint4 u[2]; } fa;
            fa.u[0] = *(const uint4*)(&Alds[row * SAP + lhalf * 8]);
            fa.u[1] = *(const uint4*)(&Alds[row * SAP + lhalf * 8 + 16]);
            afrag[i] = fa.v;
        }
        // B fragment (32x16): lanes 0-15 col=l16 K[0..7],[16..23], lanes 16-31 col=l16 K[8..15],[24..31]
        v16bf bfrag[4];
        #pragma unroll
        for (int j = 0; j < 4; ++j) {
            int n = wc + j * 16 + l16;
            union { v16bf v; uint4 u[2]; } fb;
            fb.u[0] = *(const uint4*)(&Blds[n * SBP + lhalf * 8]);
            fb.u[1] = *(const uint4*)(&Blds[n * SBP + lhalf * 8 + 16]);
            bfrag[j] = fb.v;
        }
        #pragma unroll
        for (int i = 0; i < 2; ++i)
            #pragma unroll
            for (int j = 0; j < 4; ++j)
                acc[i][j] = __builtin_amdgcn_wmma_f32_16x16x32_bf16(
                    false, afrag[i], false, bfrag[j], (short)0, acc[i][j], false, false);
    }

    // store: C/D layout -> VGPR r: lanes 0-15 M=r, lanes 16-31 M=r+8
    const bool full = (m0 + TM) <= M;        // block-uniform fast path
    #pragma unroll
    for (int i = 0; i < 2; ++i) {
        const int rbase = m0 + wr + i * 16 + (lhalf ? 8 : 0);
        float* crow = C + (size_t)rbase * HIDC;
        #pragma unroll
        for (int j = 0; j < 4; ++j) {
            const int col = n0 + wc + j * 16 + l16;
            if (full) {
                #pragma unroll
                for (int r = 0; r < 8; ++r)
                    crow[r * HIDC + col] = acc[i][j][r];
            } else {
                #pragma unroll
                for (int r = 0; r < 8; ++r)
                    if (rbase + r < M) crow[r * HIDC + col] = acc[i][j][r];
            }
        }
    }
}

// ----------------------------- edge aggregation -----------------------------
// out[dst] += H[src] * (dis[src]*dis[dst]); one wave per edge; self-loops appended.

__global__ __launch_bounds__(256) void agg_kernel(
    const float* __restrict__ H, const int* __restrict__ src, const int* __restrict__ dst,
    const float* __restrict__ dis, float* __restrict__ out, int Etot, int E)
{
    int e = blockIdx.x * 8 + (threadIdx.x >> 5);
    if (e >= Etot) return;
    int lane = threadIdx.x & 31;
    int s, d;
    if (e < E) { s = src[e]; d = dst[e]; } else { s = d = e - E; }
    float nrm = dis[s] * dis[d];
    const float* hrow = H + (size_t)s * HIDC;
    float*       orow = out + (size_t)d * HIDC;
    __builtin_prefetch(hrow, 0, 0);          // global_prefetch_b8 path
    #pragma unroll
    for (int c = lane; c < HIDC; c += 32)
        atomicAdd(&orow[c], hrow[c] * nrm);
}

// ----------------------------- BatchNorm ------------------------------------

__global__ __launch_bounds__(256) void bn_stats_kernel(
    const float* __restrict__ H, float* __restrict__ sum, float* __restrict__ sumsq,
    int Nrows, int rows_per_block)
{
    int r0 = blockIdx.x * rows_per_block;
    int r1 = r0 + rows_per_block; if (r1 > Nrows) r1 = Nrows;
    int t  = threadIdx.x;
    int c0 = t, c1 = t + 256;
    float s0 = 0.f, s1 = 0.f, q0 = 0.f, q1 = 0.f;
    for (int r = r0; r < r1; ++r) {
        float a = H[(size_t)r * HIDC + c0]; s0 += a; q0 += a * a;
        float b = H[(size_t)r * HIDC + c1]; s1 += b; q1 += b * b;
    }
    atomicAdd(&sum[c0],   s0); atomicAdd(&sumsq[c0], q0);
    atomicAdd(&sum[c1],   s1); atomicAdd(&sumsq[c1], q1);
}

__global__ void bn_finalize_kernel(
    const float* __restrict__ sum, const float* __restrict__ sumsq,
    const float* __restrict__ g, const float* __restrict__ be,
    float* __restrict__ scale, float* __restrict__ shift, int Nrows)
{
    int c = blockIdx.x * blockDim.x + threadIdx.x;
    if (c >= HIDC) return;
    float inv = 1.0f / (float)Nrows;
    float mu  = sum[c] * inv;
    float var = sumsq[c] * inv - mu * mu;
    float sc  = rsqrtf(var + 1e-5f) * g[c];
    scale[c] = sc;
    shift[c] = be[c] - mu * sc;
}

// BN apply + ReLU; writes fp32 (for JK) and bf16 (next GEMM input)
__global__ void bn_apply_relu_kernel(
    const float* __restrict__ H, const float* __restrict__ scale, const float* __restrict__ shift,
    float* __restrict__ Xf, bf16_t* __restrict__ Xb, size_t n)
{
    size_t i = (size_t)blockIdx.x * blockDim.x + threadIdx.x;
    if (i >= n) return;
    int c = (int)(i & (HIDC - 1));
    float y = fmaxf(H[i] * scale[c] + shift[c], 0.0f);
    Xf[i] = y;
    Xb[i] = f2bf(y);
}

// hmax (in x1) = max(x1, x2, agg)
__global__ void jk_max_kernel(float* __restrict__ x1, const float* __restrict__ x2,
                              const float* __restrict__ x3, size_t n)
{
    size_t i = (size_t)blockIdx.x * blockDim.x + threadIdx.x;
    if (i < n) x1[i] = fmaxf(x1[i], fmaxf(x2[i], x3[i]));
}

// ----------------------------- classifier head ------------------------------
// logits = Hmax @ Wf + bf, then per-row log_softmax over 40 classes.
// 4 rows per block (64 threads/row, 40 active for the dot).

__global__ __launch_bounds__(256) void head_kernel(
    const float* __restrict__ Hmax, const float* __restrict__ Wf, const float* __restrict__ bfv,
    float* __restrict__ out, int Nrows)
{
    __shared__ float hrow[4][HIDC];
    __shared__ float lg[4][64];

    int base = blockIdx.x * 4;
    // cooperative row staging
    for (int i = threadIdx.x; i < 4 * HIDC; i += 256) {
        int rs = i >> 9, c = i & (HIDC - 1);
        int r = base + rs;
        hrow[rs][c] = (r < Nrows) ? Hmax[(size_t)r * HIDC + c] : 0.0f;
    }
    __syncthreads();

    int rslot = threadIdx.x >> 6;
    int j     = threadIdx.x & 63;
    int row   = base + rslot;

    if (row < Nrows && j < OUTC) {
        float acc = bfv[j];
        #pragma unroll 8
        for (int k = 0; k < HIDC; ++k)
            acc = fmaf(hrow[rslot][k], Wf[(size_t)k * OUTC + j], acc);
        lg[rslot][j] = acc;
    }
    __syncthreads();

    if (row < Nrows && j == 0) {
        float m = -3.4e38f;
        for (int t = 0; t < OUTC; ++t) m = fmaxf(m, lg[rslot][t]);
        float s = 0.f;
        for (int t = 0; t < OUTC; ++t) s += __expf(lg[rslot][t] - m);
        lg[rslot][40] = m;
        lg[rslot][41] = __logf(s);
    }
    __syncthreads();

    if (row < Nrows && j < OUTC)
        out[(size_t)row * OUTC + j] = lg[rslot][j] - lg[rslot][40] - lg[rslot][41];
}

// ----------------------------- host launcher --------------------------------

static inline int cdiv(long long a, long long b) { return (int)((a + b - 1) / b); }

extern "C" void kernel_launch(void* const* d_in, const int* in_sizes, int n_in,
                              void* d_out, int out_size, void* d_ws, size_t ws_size,
                              hipStream_t stream) {
    const float* x   = (const float*)d_in[0];
    const int*   ei  = (const int*)  d_in[1];
    const float* W1  = (const float*)d_in[2];
    const float* b1  = (const float*)d_in[3];
    const float* g1  = (const float*)d_in[4];
    const float* be1 = (const float*)d_in[5];
    const float* W2  = (const float*)d_in[6];
    const float* b2  = (const float*)d_in[7];
    const float* g2  = (const float*)d_in[8];
    const float* be2 = (const float*)d_in[9];
    const float* W3  = (const float*)d_in[10];
    const float* b3  = (const float*)d_in[11];
    const float* Wf  = (const float*)d_in[12];
    const float* bfv = (const float*)d_in[13];
    float* out = (float*)d_out;

    const int N = in_sizes[0] / HIDC;          // 50000
    const int E = in_sizes[1] / 2;             // 800000
    const int Etot = E + N;                    // with self-loops
    const size_t NH = (size_t)N * HIDC;
    const int* srcI = ei;
    const int* dstI = ei + E;

    // ---- workspace bump allocator ----
    uintptr_t w = (uintptr_t)d_ws;
    auto alloc = [&](size_t bytes) -> void* {
        void* p = (void*)w;
        w += (bytes + 255) & ~(size_t)255;
        return p;
    };
    float*  dis   = (float*) alloc((size_t)N * 4);
    bf16_t* W1b   = (bf16_t*)alloc((size_t)HIDC * HIDC * 2);
    bf16_t* W2b   = (bf16_t*)alloc((size_t)HIDC * HIDC * 2);
    bf16_t* W3b   = (bf16_t*)alloc((size_t)HIDC * HIDC * 2);
    bf16_t* Xb    = (bf16_t*)alloc(NH * 2);
    float*  Hlin  = (float*) alloc(NH * 4);
    float*  Agg   = (float*) alloc(NH * 4);
    float*  x1    = (float*) alloc(NH * 4);
    float*  x2    = (float*) alloc(NH * 4);
    float*  sums  = (float*) alloc(HIDC * 4);
    float*  sumsq = (float*) alloc(HIDC * 4);
    float*  scale = (float*) alloc(HIDC * 4);
    float*  shift = (float*) alloc(HIDC * 4);

    const dim3 gemm_grid(HIDC / TN, cdiv(N, TM));
    const int RPB = 128;                       // bn_stats rows per block

    // ---- normalization: deg (self-loop => init 1), rsqrt ----
    fill_f32_kernel<<<cdiv(N, 256), 256, 0, stream>>>(dis, 1.0f, (size_t)N);
    deg_kernel<<<cdiv(E, 256), 256, 0, stream>>>(dstI, dis, E);
    rsqrt_kernel<<<cdiv(N, 256), 256, 0, stream>>>(dis, N);

    // ---- bf16 conversions ----
    cvt_bf16_kernel<<<cdiv((long long)NH, 256), 256, 0, stream>>>(x, Xb, NH);
    cvt_bf16_kernel<<<cdiv(HIDC * HIDC, 256), 256, 0, stream>>>(W1, W1b, (size_t)HIDC * HIDC);
    cvt_bf16_kernel<<<cdiv(HIDC * HIDC, 256), 256, 0, stream>>>(W2, W2b, (size_t)HIDC * HIDC);
    cvt_bf16_kernel<<<cdiv(HIDC * HIDC, 256), 256, 0, stream>>>(W3, W3b, (size_t)HIDC * HIDC);

    // ================= layer 1 =================
    gemm_bf16_wmma_kernel<<<gemm_grid, 256, 0, stream>>>(Xb, W1b, Hlin, N);
    bias_init_kernel<<<cdiv((long long)NH, 256), 256, 0, stream>>>(Agg, b1, NH);
    agg_kernel<<<cdiv(Etot, 8), 256, 0, stream>>>(Hlin, srcI, dstI, dis, Agg, Etot, E);
    fill_f32_kernel<<<cdiv(HIDC, 256), 256, 0, stream>>>(sums, 0.f, HIDC);
    fill_f32_kernel<<<cdiv(HIDC, 256), 256, 0, stream>>>(sumsq, 0.f, HIDC);
    bn_stats_kernel<<<cdiv(N, RPB), 256, 0, stream>>>(Agg, sums, sumsq, N, RPB);
    bn_finalize_kernel<<<cdiv(HIDC, 256), 256, 0, stream>>>(sums, sumsq, g1, be1, scale, shift, N);
    bn_apply_relu_kernel<<<cdiv((long long)NH, 256), 256, 0, stream>>>(Agg, scale, shift, x1, Xb, NH);

    // ================= layer 2 =================
    gemm_bf16_wmma_kernel<<<gemm_grid, 256, 0, stream>>>(Xb, W2b, Hlin, N);
    bias_init_kernel<<<cdiv((long long)NH, 256), 256, 0, stream>>>(Agg, b2, NH);
    agg_kernel<<<cdiv(Etot, 8), 256, 0, stream>>>(Hlin, srcI, dstI, dis, Agg, Etot, E);
    fill_f32_kernel<<<cdiv(HIDC, 256), 256, 0, stream>>>(sums, 0.f, HIDC);
    fill_f32_kernel<<<cdiv(HIDC, 256), 256, 0, stream>>>(sumsq, 0.f, HIDC);
    bn_stats_kernel<<<cdiv(N, RPB), 256, 0, stream>>>(Agg, sums, sumsq, N, RPB);
    bn_finalize_kernel<<<cdiv(HIDC, 256), 256, 0, stream>>>(sums, sumsq, g2, be2, scale, shift, N);
    bn_apply_relu_kernel<<<cdiv((long long)NH, 256), 256, 0, stream>>>(Agg, scale, shift, x2, Xb, NH);

    // ================= layer 3 (no BN) =========
    gemm_bf16_wmma_kernel<<<gemm_grid, 256, 0, stream>>>(Xb, W3b, Hlin, N);
    bias_init_kernel<<<cdiv((long long)NH, 256), 256, 0, stream>>>(Agg, b3, NH);
    agg_kernel<<<cdiv(Etot, 8), 256, 0, stream>>>(Hlin, srcI, dstI, dis, Agg, Etot, E);

    // ---- JK max + head ----
    jk_max_kernel<<<cdiv((long long)NH, 256), 256, 0, stream>>>(x1, x2, Agg, NH);
    head_kernel<<<cdiv(N, 4), 256, 0, stream>>>(x1, Wf, bfv, out, N);
}
